// SparseTransformerEncoderLayer_9929964388667
// MI455X (gfx1250) — matrix operand order repaired and checked by reading, not compiled, not run on hardware
//
#include <hip/hip_runtime.h>
#include <hip/hip_bf16.h>
#include <math.h>

// Problem dims (match reference)
#define BQ    4
#define SEQ   2048
#define DM    512
#define NH    8
#define HDIM  64
#define NBLK  32          // SEQ / 64 key/query blocks
#define DFFN  2048
#define MROWS (BQ*SEQ)    // 8192 token rows

typedef __attribute__((ext_vector_type(16))) __bf16 v16bf;
typedef __attribute__((ext_vector_type(8)))  float  v8f;

union FragBF { v16bf v; unsigned int u[8]; };

__device__ __forceinline__ unsigned short f2bf(float f) {
  union { float f; unsigned int u; } c; c.f = f;
  return (unsigned short)((c.u + 0x7FFFu + ((c.u >> 16) & 1u)) >> 16);
}

// ---------------------------------------------------------------------------
// CDNA5 async global->LDS copy (ASYNCcnt path, ISA 15.18.3 op 98).
// LDS operand is the raw LDS byte offset (VGPR); global address is a 64-bit
// per-lane VGPR pair, "off" (no SADDR).
// ---------------------------------------------------------------------------
__device__ __forceinline__ unsigned lds_off(const void* p) {
  return (unsigned)(size_t)(const __attribute__((address_space(3))) void*)p;
}
__device__ __forceinline__ void async_copy_b128(unsigned lds, const void* gptr) {
  asm volatile("global_load_async_to_lds_b128 %0, %1, off"
               :: "v"(lds), "v"((unsigned long long)(size_t)gptr)
               : "memory");
}
__device__ __forceinline__ void wait_async(int pending) {
  if (pending) asm volatile("s_wait_asynccnt 0x4" ::: "memory");
  else         asm volatile("s_wait_asynccnt 0x0" ::: "memory");
}

// ---------------------------------------------------------------------------
// f32 -> bf16 (raw bits) conversion
// ---------------------------------------------------------------------------
__global__ void cvt_f32_bf16(const float* __restrict__ in,
                             unsigned short* __restrict__ out, int n) {
  for (int i = blockIdx.x * blockDim.x + threadIdx.x; i < n;
       i += gridDim.x * blockDim.x)
    out[i] = f2bf(in[i]);
}

// ---------------------------------------------------------------------------
// Block-sparse mask -> compact per-qblock key-block lists
// ---------------------------------------------------------------------------
__global__ void build_mask(int* __restrict__ klist, int* __restrict__ kcount) {
  const int qb = threadIdx.x;
  if (qb >= NBLK) return;
  bool m[NBLK];
  for (int kb = 0; kb < NBLK; ++kb) {
    int d = qb - kb; if (d < 0) d = -d;
    m[kb] = (qb == 0) || (kb == 0) || (d <= 1);
  }
  unsigned int st = 0x9E3779B9u ^ (0x85EBCA6Bu * (unsigned)qb);
  for (int j = 0; j < 2; ++j) {
    st = st * 1664525u + 1013904223u;
    m[(st >> 11) & 31] = true;
  }
  int c = 0;
  for (int kb = 0; kb < NBLK; ++kb)
    if (m[kb]) klist[qb * NBLK + (c++)] = kb;
  kcount[qb] = c;
}

// ---------------------------------------------------------------------------
// Shared 64x64 GEMM tile core with async double-buffered K panels.
// blockDim=128 (4 waves); wave w owns 16-col strip w, acc[4] of v8f.
// A: [M x K] bf16 row-major, W: [N x K] bf16 row-major (B = W^T).
// ---------------------------------------------------------------------------
#define LPAD 8
#define LSTR (32 + LPAD)
#define PANEL (64 * LSTR)

__device__ __forceinline__ void stage_panel_async(
    const unsigned short* __restrict__ A, int lda, int rowBase,
    const unsigned short* __restrict__ W, int ldw, int colBase, int k0,
    unsigned short* bufA, unsigned short* bufW, int tid) {
  for (int it = tid; it < 256; it += 128) {      // 64 rows x 32 bf16, 16B each
    int r = it >> 2, c = (it & 3) << 3;
    async_copy_b128(lds_off(bufA + r * LSTR + c),
                    A + (size_t)(rowBase + r) * lda + k0 + c);
    async_copy_b128(lds_off(bufW + r * LSTR + c),
                    W + (size_t)(colBase + r) * ldw + k0 + c);
  }
}

__device__ __forceinline__ void gemm_tile_64x64(
    const unsigned short* __restrict__ A, int lda,
    const unsigned short* __restrict__ W, int ldw,
    int K, int rowBase, int colBase,
    unsigned short* ldsA, unsigned short* ldsW, v8f acc[4]) {
  const int tid  = threadIdx.x;
  const int lane = tid & 31;
  const int wv   = tid >> 5;
  const int c16  = lane & 15;
  const int hf   = lane >> 4;   // K-half select per ISA 16-bit layouts
  stage_panel_async(A, lda, rowBase, W, ldw, colBase, 0, ldsA, ldsW, tid);
  int buf = 0;
  for (int k0 = 0; k0 < K; k0 += 32) {
    unsigned short* curA = ldsA + buf * PANEL;
    unsigned short* curW = ldsW + buf * PANEL;
    const int has_next = (k0 + 32 < K);
    if (has_next)   // stream next panel while computing this one
      stage_panel_async(A, lda, rowBase, W, ldw, colBase, k0 + 32,
                        ldsA + (buf ^ 1) * PANEL, ldsW + (buf ^ 1) * PANEL, tid);
    wait_async(has_next);   // async loads retire in order: current panel landed
    __syncthreads();
    FragBF bfr;   // B frag: lane=N col, K = (v<4 ? 2v : 16+2(v-4)) + hf*8
    {
      const unsigned short* wrow = curW + (wv * 16 + c16) * LSTR + hf * 8;
#pragma unroll
      for (int v = 0; v < 8; ++v)
        bfr.u[v] = *(const unsigned int*)(wrow + ((v < 4) ? (2 * v) : (16 + 2 * (v - 4))));
    }
#pragma unroll
    for (int tr = 0; tr < 4; ++tr) {
      FragBF afr;  // A frag: lane=M row, K = (v<4 ? 2v : 16+2(v-4)) + hf*8
      const unsigned short* arow = curA + (tr * 16 + c16) * LSTR + hf * 8;
#pragma unroll
      for (int v = 0; v < 8; ++v) {
        int kk = (v < 4) ? (2 * v) : (16 + 2 * (v - 4));
        afr.u[v] = *(const unsigned int*)(arow + kk);
      }
      acc[tr] = __builtin_amdgcn_wmma_f32_16x16x32_bf16(
          false, afr.v, false, bfr.v, (short)0, acc[tr], false, false);
    }
    __syncthreads();
    buf ^= 1;
  }
}

// ---------------------------------------------------------------------------
// QKV projection: qkv = x @ Wqkv^T + b, scattered into q/k/v buffers
// ---------------------------------------------------------------------------
__global__ __launch_bounds__(128)
void qkv_gemm(const unsigned short* __restrict__ xbf,
              const unsigned short* __restrict__ wbf,
              const float* __restrict__ bias,
              unsigned short* __restrict__ qo,
              unsigned short* __restrict__ ko,
              unsigned short* __restrict__ vo) {
  __shared__ unsigned short ldsA[2 * PANEL];
  __shared__ unsigned short ldsW[2 * PANEL];
  const int rowBase = blockIdx.y * 64, colBase = blockIdx.x * 64;
  const v8f vzero = {0.f, 0.f, 0.f, 0.f, 0.f, 0.f, 0.f, 0.f};
  v8f acc[4];
#pragma unroll
  for (int i = 0; i < 4; ++i) acc[i] = vzero;
  gemm_tile_64x64(xbf, DM, wbf, DM, DM, rowBase, colBase, ldsA, ldsW, acc);
  const int tid = threadIdx.x, lane = tid & 31, wv = tid >> 5;
  const int c16 = lane & 15, hf = lane >> 4;
  const int n = colBase + wv * 16 + c16;
  const float bv = bias[n];
  unsigned short* dst; int nOff;                 // 64-col tile never straddles
  if (n < DM)          { dst = qo; nOff = n; }
  else if (n < 2 * DM) { dst = ko; nOff = n - DM; }
  else                 { dst = vo; nOff = n - 2 * DM; }
#pragma unroll
  for (int tr = 0; tr < 4; ++tr)
#pragma unroll
    for (int vv = 0; vv < 8; ++vv) {
      int m = rowBase + tr * 16 + 8 * hf + vv;
      dst[(size_t)m * DM + nOff] = f2bf(acc[tr][vv] + bv);
    }
}

// ---------------------------------------------------------------------------
// GEMM + bias + residual -> f32 (out-proj and FFN down-proj)
// ---------------------------------------------------------------------------
__global__ __launch_bounds__(128)
void proj_add_gemm(const unsigned short* __restrict__ A,
                   const unsigned short* __restrict__ W,
                   const float* __restrict__ bias,
                   const float* __restrict__ resid,
                   float* __restrict__ outp, int N, int K) {
  __shared__ unsigned short ldsA[2 * PANEL];
  __shared__ unsigned short ldsW[2 * PANEL];
  const int rowBase = blockIdx.y * 64, colBase = blockIdx.x * 64;
  const v8f vzero = {0.f, 0.f, 0.f, 0.f, 0.f, 0.f, 0.f, 0.f};
  v8f acc[4];
#pragma unroll
  for (int i = 0; i < 4; ++i) acc[i] = vzero;
  gemm_tile_64x64(A, K, W, K, K, rowBase, colBase, ldsA, ldsW, acc);
  const int tid = threadIdx.x, lane = tid & 31, wv = tid >> 5;
  const int c16 = lane & 15, hf = lane >> 4;
  const int n = colBase + wv * 16 + c16;
  const float bv = bias[n];
#pragma unroll
  for (int tr = 0; tr < 4; ++tr)
#pragma unroll
    for (int vv = 0; vv < 8; ++vv) {
      int m = rowBase + tr * 16 + 8 * hf + vv;
      outp[(size_t)m * N + n] = resid[(size_t)m * N + n] + acc[tr][vv] + bv;
    }
}

// ---------------------------------------------------------------------------
// FFN up-proj: relu(y @ w1^T + b1) -> bf16
// ---------------------------------------------------------------------------
__global__ __launch_bounds__(128)
void ffn1_gemm(const unsigned short* __restrict__ A,
               const unsigned short* __restrict__ W,
               const float* __restrict__ bias,
               unsigned short* __restrict__ outp) {
  __shared__ unsigned short ldsA[2 * PANEL];
  __shared__ unsigned short ldsW[2 * PANEL];
  const int rowBase = blockIdx.y * 64, colBase = blockIdx.x * 64;
  const v8f vzero = {0.f, 0.f, 0.f, 0.f, 0.f, 0.f, 0.f, 0.f};
  v8f acc[4];
#pragma unroll
  for (int i = 0; i < 4; ++i) acc[i] = vzero;
  gemm_tile_64x64(A, DM, W, DM, DM, rowBase, colBase, ldsA, ldsW, acc);
  const int tid = threadIdx.x, lane = tid & 31, wv = tid >> 5;
  const int c16 = lane & 15, hf = lane >> 4;
  const int n = colBase + wv * 16 + c16;
  const float bv = bias[n];
#pragma unroll
  for (int tr = 0; tr < 4; ++tr)
#pragma unroll
    for (int vv = 0; vv < 8; ++vv) {
      int m = rowBase + tr * 16 + 8 * hf + vv;
      outp[(size_t)m * DFFN + n] = f2bf(fmaxf(acc[tr][vv] + bv, 0.f));
    }
}

// ---------------------------------------------------------------------------
// Block-sparse flash attention over 64x64 tiles, per (b, h, qblock).
// K/V tiles streamed via async global->LDS; V staged row-major into Ps and
// transposed LDS->LDS into Vt (Ps softmax overwrite happens after the next
// barrier, so the reuse is race-free).
// ---------------------------------------------------------------------------
__global__ __launch_bounds__(128)
void attention_kernel(const unsigned short* __restrict__ q,
                      const unsigned short* __restrict__ k,
                      const unsigned short* __restrict__ v,
                      unsigned short* __restrict__ ctx,
                      const int* __restrict__ klist,
                      const int* __restrict__ kcount) {
  __shared__ unsigned short Qs[64 * 72];
  __shared__ unsigned short Ks[64 * 72];
  __shared__ unsigned short Vt[64 * 72];   // transposed: Vt[d][key]
  __shared__ unsigned short Ps[64 * 72];   // V staging, then softmax probs
  __shared__ float Ss[64 * 68];
  __shared__ float mrow[64], lrow[64], arow[64];
  const int qb = blockIdx.x, h = blockIdx.y, b = blockIdx.z;
  const size_t base = ((size_t)b * SEQ) * DM + (size_t)h * HDIM;
  const int tid = threadIdx.x, lane = tid & 31, wv = tid >> 5;
  const int c16 = lane & 15, hf = lane >> 4;

  for (int it = tid; it < 512; it += 128) {
    int r = it >> 3, c = (it & 7) << 3;
    async_copy_b128(lds_off(Qs + r * 72 + c),
                    q + base + (size_t)(qb * 64 + r) * DM + c);
  }
  if (tid < 64) { mrow[tid] = -1e30f; lrow[tid] = 0.f; }
  const v8f vzero = {0.f, 0.f, 0.f, 0.f, 0.f, 0.f, 0.f, 0.f};
  v8f acc[4];
#pragma unroll
  for (int i = 0; i < 4; ++i) acc[i] = vzero;
  wait_async(0);
  __syncthreads();

  const int cnt = kcount[qb];
  for (int ib = 0; ib < cnt; ++ib) {
    const int kb = klist[qb * NBLK + ib];
    for (int it = tid; it < 512; it += 128) {
      int r = it >> 3, c = (it & 7) << 3;
      async_copy_b128(lds_off(Ks + r * 72 + c),
                      k + base + (size_t)(kb * 64 + r) * DM + c);
      async_copy_b128(lds_off(Ps + r * 72 + c),      // V rows, staged
                      v + base + (size_t)(kb * 64 + r) * DM + c);
    }
    wait_async(0);
    __syncthreads();

    // LDS transpose of staged V rows -> Vt[d][key]
    for (int it = tid; it < 4096; it += 128) {
      int r = it >> 6, c = it & 63;
      Vt[c * 72 + r] = Ps[r * 72 + c];
    }

    // S = Q @ K^T   (B-matrix cols = key rows; pairs over d contiguous in Ks)
    v8f sc[4];
#pragma unroll
    for (int i = 0; i < 4; ++i) sc[i] = vzero;
#pragma unroll
    for (int k0 = 0; k0 < 64; k0 += 32) {
      FragBF bfr;
      const unsigned short* krow = Ks + (wv * 16 + c16) * 72 + k0 + hf * 8;
#pragma unroll
      for (int vv = 0; vv < 8; ++vv)
        bfr.u[vv] = *(const unsigned int*)(krow + ((vv < 4) ? (2 * vv) : (16 + 2 * (vv - 4))));
#pragma unroll
      for (int tr = 0; tr < 4; ++tr) {
        FragBF afr;
        const unsigned short* qrow = Qs + (tr * 16 + c16) * 72 + k0 + hf * 8;
#pragma unroll
        for (int vv = 0; vv < 8; ++vv) {
          int kk = (vv < 4) ? (2 * vv) : (16 + 2 * (vv - 4));
          afr.u[vv] = *(const unsigned int*)(qrow + kk);
        }
        sc[tr] = __builtin_amdgcn_wmma_f32_16x16x32_bf16(
            false, afr.v, false, bfr.v, (short)0, sc[tr], false, false);
      }
    }
    const float sscale = 0.125f;   // 1/sqrt(HD=64)
#pragma unroll
    for (int tr = 0; tr < 4; ++tr)
#pragma unroll
      for (int vv = 0; vv < 8; ++vv)
        Ss[(tr * 16 + 8 * hf + vv) * 68 + wv * 16 + c16] = sc[tr][vv] * sscale;
    __syncthreads();   // Ss ready; all transpose reads of Ps done

    // online softmax: thread t owns query row t (overwrites Ps with probs)
    if (tid < 64) {
      float* srow = Ss + tid * 68;
      float mb = srow[0];
      for (int i = 1; i < 64; ++i) mb = fmaxf(mb, srow[i]);
      float mp = mrow[tid];
      float mn = fmaxf(mp, mb);
      float al = __expf(mp - mn);
      float sum = 0.f;
      unsigned short* prow = Ps + tid * 72;
      for (int i = 0; i < 64; ++i) {
        float p = __expf(srow[i] - mn);
        sum += p;
        prow[i] = f2bf(p);
      }
      lrow[tid] = lrow[tid] * al + sum;
      mrow[tid] = mn;
      arow[tid] = al;
    }
    __syncthreads();

    // ctx = ctx*alpha + P @ V
#pragma unroll
    for (int tr = 0; tr < 4; ++tr)
#pragma unroll
      for (int vv = 0; vv < 8; ++vv)
        acc[tr][vv] *= arow[tr * 16 + 8 * hf + vv];
#pragma unroll
    for (int k0 = 0; k0 < 64; k0 += 32) {
      FragBF bfr;
      const unsigned short* vrow = Vt + (wv * 16 + c16) * 72 + k0 + hf * 8;
#pragma unroll
      for (int vv = 0; vv < 8; ++vv)
        bfr.u[vv] = *(const unsigned int*)(vrow + ((vv < 4) ? (2 * vv) : (16 + 2 * (vv - 4))));
#pragma unroll
      for (int tr = 0; tr < 4; ++tr) {
        FragBF afr;
        const unsigned short* prow2 = Ps + (tr * 16 + c16) * 72 + k0 + hf * 8;
#pragma unroll
        for (int vv = 0; vv < 8; ++vv) {
          int kk = (vv < 4) ? (2 * vv) : (16 + 2 * (vv - 4));
          afr.u[vv] = *(const unsigned int*)(prow2 + kk);
        }
        acc[tr] = __builtin_amdgcn_wmma_f32_16x16x32_bf16(
            false, afr.v, false, bfr.v, (short)0, acc[tr], false, false);
      }
    }
    __syncthreads();
  }
#pragma unroll
  for (int tr = 0; tr < 4; ++tr)
#pragma unroll
    for (int vv = 0; vv < 8; ++vv) {
      int r = tr * 16 + 8 * hf + vv;
      float val = acc[tr][vv] / lrow[r];
      ctx[base + (size_t)(qb * 64 + r) * DM + wv * 16 + c16] = f2bf(val);
    }
}

// ---------------------------------------------------------------------------
// LayerNorm: one wave32 per 512-elem row; shuffle reductions
// ---------------------------------------------------------------------------
__global__ __launch_bounds__(256)
void ln_kernel(const float* __restrict__ in, const float* __restrict__ g,
               const float* __restrict__ bb, float* __restrict__ outp,
               unsigned short* __restrict__ outbf) {
  const int gw = (blockIdx.x * blockDim.x + threadIdx.x) >> 5;
  const int lane = threadIdx.x & 31;
  if (gw >= MROWS) return;
  const float* row = in + (size_t)gw * DM;
  float s = 0.f, s2 = 0.f;
  for (int i = lane; i < DM; i += 32) { float t = row[i]; s += t; s2 += t * t; }
#pragma unroll
  for (int o = 16; o > 0; o >>= 1) {
    s  += __shfl_xor(s, o, 32);
    s2 += __shfl_xor(s2, o, 32);
  }
  const float mu = s * (1.f / DM);
  const float inv = rsqrtf(s2 * (1.f / DM) - mu * mu + 1e-5f);
  for (int i = lane; i < DM; i += 32) {
    float t = (row[i] - mu) * inv * g[i] + bb[i];
    outp[(size_t)gw * DM + i] = t;
    if (outbf) outbf[(size_t)gw * DM + i] = f2bf(t);
  }
}

// ---------------------------------------------------------------------------
extern "C" void kernel_launch(void* const* d_in, const int* in_sizes, int n_in,
                              void* d_out, int out_size, void* d_ws,
                              size_t ws_size, hipStream_t stream) {
  const float* x    = (const float*)d_in[0];
  const float* wqkv = (const float*)d_in[1];
  const float* bqkv = (const float*)d_in[2];
  const float* wout = (const float*)d_in[3];
  const float* bout = (const float*)d_in[4];
  const float* w1   = (const float*)d_in[5];
  const float* b1   = (const float*)d_in[6];
  const float* w2   = (const float*)d_in[7];
  const float* b2   = (const float*)d_in[8];
  const float* g1   = (const float*)d_in[9];
  const float* be1  = (const float*)d_in[10];
  const float* g2   = (const float*)d_in[11];
  const float* be2  = (const float*)d_in[12];
  float* out = (float*)d_out;

  char* ws = (char*)d_ws;
  size_t off = 0;
  auto carve = [&](size_t bytes) -> char* {
    char* p = ws + off;
    off += (bytes + 255) & ~(size_t)255;
    return p;
  };
  unsigned short* xbf   = (unsigned short*)carve((size_t)MROWS * DM * 2);
  unsigned short* wqkvb = (unsigned short*)carve((size_t)3 * DM * DM * 2);
  unsigned short* woutb = (unsigned short*)carve((size_t)DM * DM * 2);
  unsigned short* w1b   = (unsigned short*)carve((size_t)DFFN * DM * 2);
  unsigned short* w2b   = (unsigned short*)carve((size_t)DM * DFFN * 2);
  unsigned short* qb_   = (unsigned short*)carve((size_t)MROWS * DM * 2);
  unsigned short* kb_   = (unsigned short*)carve((size_t)MROWS * DM * 2);
  unsigned short* vb_   = (unsigned short*)carve((size_t)MROWS * DM * 2);
  unsigned short* ctxb  = (unsigned short*)carve((size_t)MROWS * DM * 2);
  unsigned short* ybf   = (unsigned short*)carve((size_t)MROWS * DM * 2);
  unsigned short* hb    = (unsigned short*)carve((size_t)MROWS * DFFN * 2);
  float* tmp1 = (float*)carve((size_t)MROWS * DM * 4);
  float* yf   = (float*)carve((size_t)MROWS * DM * 4);
  int* klist  = (int*)carve((size_t)NBLK * NBLK * 4);
  int* kcount = (int*)carve((size_t)NBLK * 4);

  // 0) mask + bf16 staging
  build_mask<<<1, 32, 0, stream>>>(klist, kcount);
  cvt_f32_bf16<<<1024, 256, 0, stream>>>(x, xbf, MROWS * DM);
  cvt_f32_bf16<<<512, 256, 0, stream>>>(wqkv, wqkvb, 3 * DM * DM);
  cvt_f32_bf16<<<256, 256, 0, stream>>>(wout, woutb, DM * DM);
  cvt_f32_bf16<<<512, 256, 0, stream>>>(w1, w1b, DFFN * DM);
  cvt_f32_bf16<<<512, 256, 0, stream>>>(w2, w2b, DM * DFFN);

  // 1) QKV projection
  qkv_gemm<<<dim3(3 * DM / 64, MROWS / 64), 128, 0, stream>>>(
      xbf, wqkvb, bqkv, qb_, kb_, vb_);

  // 2) block-sparse flash attention
  attention_kernel<<<dim3(NBLK, NH, BQ), 128, 0, stream>>>(
      qb_, kb_, vb_, ctxb, klist, kcount);

  // 3) out-proj + residual -> LN1
  proj_add_gemm<<<dim3(DM / 64, MROWS / 64), 128, 0, stream>>>(
      ctxb, woutb, bout, x, tmp1, DM, DM);
  ln_kernel<<<MROWS / 8, 256, 0, stream>>>(tmp1, g1, be1, yf, ybf);

  // 4) FFN
  ffn1_gemm<<<dim3(DFFN / 64, MROWS / 64), 128, 0, stream>>>(ybf, w1b, b1, hb);
  proj_add_gemm<<<dim3(DM / 64, MROWS / 64), 128, 0, stream>>>(
      hb, w2b, b2, yf, tmp1, DM, DFFN);
  ln_kernel<<<MROWS / 8, 256, 0, stream>>>(tmp1, g2, be2, out, nullptr);
}
